// RSSM_86019605004562
// MI455X (gfx1250) — matrix operand order repaired and checked by reading, not compiled, not run on hardware
//
#include <hip/hip_runtime.h>
#include <hip/hip_bf16.h>

// ---------------------------------------------------------------------------
// GRU rollout (RSSM) for MI455X / gfx1250, wave32, bf16 WMMA path.
// B=256, L=1024, LAT=256, ACT=6, D=256, GIN=518, 3D=768.
// ---------------------------------------------------------------------------

typedef __attribute__((ext_vector_type(16))) __bf16 v16bf;
typedef __attribute__((ext_vector_type(8)))  float  v8f;

#define Bsz   256
#define Lseq  1024
#define Dh    256
#define G3    768           // 3*D
#define ACTW  6
#define WIH_STRIDE 576      // packed bf16 w_ih: [768][576] : z 0..255 | act 256..261 | pad | goal 288..543 | pad
#define WHH_STRIDE 256      // packed bf16 w_hh: [768][256]

// LDS strides (ushort / float units) padded for 4-bank skew (conflict-free b128 column loads)
#define XZ_STR  264
#define XA_STR  40
#define F32_STR 260
#define PRE_STR 516

__device__ __forceinline__ unsigned short f2bf(float f) {
    unsigned u = __float_as_uint(f);
    unsigned r = u + 0x7FFFu + ((u >> 16) & 1u);   // round-to-nearest-even
    return (unsigned short)(r >> 16);
}

__device__ __forceinline__ v8f wmma_bf16(v16bf a, v16bf b, v8f c) {
    return __builtin_amdgcn_wmma_f32_16x16x32_bf16(
        /*neg_a=*/false, a, /*neg_b=*/false, b,
        /*c_mod=*/(short)0, c, /*reuse_a=*/false, /*reuse_b=*/false);
}

// A fragment, 16x32 bf16, from row-major LDS tile.
// Lanes 0-15: rows 0-15, K = k0..k0+7 then k0+16..k0+23.
// Lanes 16-31: rows 0-15, K = k0+8..k0+15 then k0+24..k0+31.
__device__ __forceinline__ v16bf load_fragA(const unsigned short* base, int stride,
                                            int lane, int k0) {
    int row  = lane & 15;
    int half = lane >> 4;
    const unsigned short* p = base + row * stride + k0 + half * 8;
    union { v16bf v; uint4 q[2]; } u;
    u.q[0] = *(const uint4*)(p);
    u.q[1] = *(const uint4*)(p + 16);
    return u.v;
}

// B fragment, 32x16 bf16 (KxN), from row-major bf16 weights w[n][k] (K contiguous per column).
// Lanes 0-15: col N=n0+lane, K=k0..k0+7 then k0+16..k0+23.  Lanes 16-31: col N=n0+lane-16, K=k0+8..k0+15 then k0+24..k0+31.
__device__ __forceinline__ v16bf load_fragB(const unsigned short* w, int stride,
                                            int n0, int lane, int k0) {
    int n  = n0 + (lane & 15);
    int kh = lane >> 4;
    const unsigned short* p = w + (size_t)n * stride + k0 + kh * 8;
    union { v16bf v; uint4 q[2]; } u;
    u.q[0] = *(const uint4*)(p);
    u.q[1] = *(const uint4*)(p + 16);
    return u.v;
}

__device__ __forceinline__ v8f bias_bcast(const float* b, int n0, int lane) {
    float v = b[n0 + (lane & 15)];
    v8f c;
#pragma unroll
    for (int i = 0; i < 8; ++i) c[i] = v;
    return c;
}

__device__ __forceinline__ float sigmoidf_(float x) {
    return 1.0f / (1.0f + __expf(-x));
}

// ---------------------------------------------------------------------------
// Kernel 0: convert/pack weights f32 -> bf16 (row-major, K-padded).
// ---------------------------------------------------------------------------
__global__ void pack_weights_kernel(const float* __restrict__ w_ih,
                                    const float* __restrict__ w_hh,
                                    unsigned short* __restrict__ wihb,
                                    unsigned short* __restrict__ whhb) {
    int n = blockIdx.x;                       // 0..767 (output unit)
    for (int c = threadIdx.x; c < WIH_STRIDE; c += blockDim.x) {
        float v = 0.0f;
        if (c < 262)                 v = w_ih[(size_t)n * 518 + c];        // z(0..255) + act(256..261)
        else if (c >= 288 && c < 544) v = w_ih[(size_t)n * 518 + (c - 26)]; // goal src 262..517
        wihb[(size_t)n * WIH_STRIDE + c] = f2bf(v);
    }
    for (int c = threadIdx.x; c < WHH_STRIDE; c += blockDim.x) {
        whhb[(size_t)n * WHH_STRIDE + c] = f2bf(w_hh[(size_t)n * 256 + c]);
    }
}

// ---------------------------------------------------------------------------
// Kernel 1: persistent fused GRU scan. 16 blocks x 512 threads (16 waves).
// Block bg owns batch rows [bg*16, bg*16+16) for all 1024 timesteps.
// Wave w owns output columns [w*48, w*48+48) (3 WMMA N-tiles).
// ---------------------------------------------------------------------------
__global__ void __launch_bounds__(512, 1)
gru_scan_kernel(const float* __restrict__ z_seq,
                const float* __restrict__ act_seq,
                const float* __restrict__ goal_proj,
                const float* __restrict__ h_init,
                const float* __restrict__ b_ih,
                const float* __restrict__ b_hh,
                const unsigned short* __restrict__ wihb,
                const unsigned short* __restrict__ whhb,
                float* __restrict__ out) {
    extern __shared__ char smem[];
    unsigned short* XZ = (unsigned short*)smem;          // 16 x XZ_STR bf16 (z tile / goal at init)
    unsigned short* XA = XZ + 16 * XZ_STR;               // 16 x XA_STR bf16 (a_prev tile, padded K=32)
    unsigned short* HB = XA + 16 * XA_STR;               // 16 x XZ_STR bf16 (h state, bf16 copy)
    float* H32 = (float*)(HB + 16 * XZ_STR);             // 16 x F32_STR f32 (h state)
    float* PRE = H32 + 16 * F32_STR;                     // 16 x PRE_STR f32 (r|u pre-activations)
    float* XN  = PRE + 16 * PRE_STR;                     // 16 x F32_STR f32 (gx, n-gate)
    float* HN  = XN  + 16 * F32_STR;                     // 16 x F32_STR f32 (gh, n-gate)

    const int tid  = threadIdx.x;
    const int lane = tid & 31;
    const int wv   = tid >> 5;          // wave 0..15
    const int b0   = blockIdx.x * 16;   // first batch row
    const int n0w  = wv * 48;           // first output column of this wave

    // ---- init: zero a-tile padding, load h_init, stage goal tile in XZ ----
    for (int i = tid; i < 16 * XA_STR; i += 512) XA[i] = 0;
    for (int i = tid; i < 16 * 256; i += 512) {
        int r = i >> 8, c = i & 255;
        float h = h_init[(size_t)(b0 + r) * Dh + c];
        H32[r * F32_STR + c] = h;
        HB[r * XZ_STR + c]   = f2bf(h);
        XZ[r * XZ_STR + c]   = f2bf(goal_proj[(size_t)(b0 + r) * Dh + c]);
    }
    __syncthreads();

    // ---- hoisted time-invariant part: gxc = goal @ W_goal^T + b_ih (regs) ----
    v8f gxc[3], bhhc[3];
#pragma unroll
    for (int tI = 0; tI < 3; ++tI) {
        int nt = n0w + 16 * tI;
        v8f c = bias_bcast(b_ih, nt, lane);
#pragma unroll
        for (int kc = 0; kc < 8; ++kc) {
            v16bf a = load_fragA(XZ, XZ_STR, lane, kc * 32);
            v16bf b = load_fragB(wihb, WIH_STRIDE, nt, lane, 288 + kc * 32);
            c = wmma_bf16(a, b, c);
        }
        gxc[tI]  = c;
        bhhc[tI] = bias_bcast(b_hh, nt, lane);
    }
    __syncthreads();   // XZ is reused for z tiles below

    // ---- sequential scan over time ----
    for (int t = 0; t < Lseq; ++t) {
        // stage z_t (f32 -> bf16) and a_{t-1}
        for (int i = tid; i < 16 * 256; i += 512) {
            int r = i >> 8, c = i & 255;
            XZ[r * XZ_STR + c] =
                f2bf(z_seq[((size_t)(b0 + r) * Lseq + t) * Dh + c]);
        }
        if (tid < 16 * ACTW) {
            int r = tid / ACTW, c = tid % ACTW;
            float a = (t == 0) ? 0.0f
                : act_seq[((size_t)(b0 + r) * Lseq + (t - 1)) * ACTW + c];
            XA[r * XA_STR + c] = f2bf(a);
        }
        __syncthreads();

        // ---- GEMMs: gx = gxc + z@Wz^T + a@Wa^T ; gh = b_hh + h@Whh^T ----
        v8f gx[3], gh[3];
#pragma unroll
        for (int tI = 0; tI < 3; ++tI) { gx[tI] = gxc[tI]; gh[tI] = bhhc[tI]; }
#pragma unroll
        for (int kc = 0; kc < 8; ++kc) {
            v16bf az = load_fragA(XZ, XZ_STR, lane, kc * 32);
            v16bf ah = load_fragA(HB, XZ_STR, lane, kc * 32);
#pragma unroll
            for (int tI = 0; tI < 3; ++tI) {
                int nt = n0w + 16 * tI;
                v16bf bz = load_fragB(wihb, WIH_STRIDE, nt, lane, kc * 32);
                gx[tI] = wmma_bf16(az, bz, gx[tI]);
                v16bf bh = load_fragB(whhb, WHH_STRIDE, nt, lane, kc * 32);
                gh[tI] = wmma_bf16(ah, bh, gh[tI]);
            }
        }
        {   // action contribution: single K=32 chunk (cols 256..287, zero-padded)
            v16bf aa = load_fragA(XA, XA_STR, lane, 0);
#pragma unroll
            for (int tI = 0; tI < 3; ++tI) {
                int nt = n0w + 16 * tI;
                v16bf ba = load_fragB(wihb, WIH_STRIDE, nt, lane, 256);
                gx[tI] = wmma_bf16(aa, ba, gx[tI]);
            }
        }

        // ---- spill accumulators: r/u tiles as (gx+gh) sum; n tiles keep both ----
#pragma unroll
        for (int tI = 0; tI < 3; ++tI) {
            int nt = n0w + 16 * tI;
            int rb = (lane >> 4) * 8;
            if (nt < 512) {
                int col = nt + (lane & 15);
#pragma unroll
                for (int r = 0; r < 8; ++r)
                    PRE[(rb + r) * PRE_STR + col] = gx[tI][r] + gh[tI][r];
            } else {
                int col = (nt - 512) + (lane & 15);
#pragma unroll
                for (int r = 0; r < 8; ++r) {
                    XN[(rb + r) * F32_STR + col] = gx[tI][r];
                    HN[(rb + r) * F32_STR + col] = gh[tI][r];
                }
            }
        }
        __syncthreads();

        // ---- gates + state update + stream h to output ----
        for (int i = tid; i < 16 * 256; i += 512) {
            int r = i >> 8, j = i & 255;
            float rg = sigmoidf_(PRE[r * PRE_STR + j]);
            float ug = sigmoidf_(PRE[r * PRE_STR + 256 + j]);
            float ng = tanhf(XN[r * F32_STR + j] + rg * HN[r * F32_STR + j]);
            float hp = H32[r * F32_STR + j];
            float hn = (1.0f - ug) * ng + ug * hp;
            H32[r * F32_STR + j] = hn;
            HB[r * XZ_STR + j]   = f2bf(hn);
            out[((size_t)(b0 + r) * Lseq + t) * Dh + j] = hn;
        }
        __syncthreads();
    }
}

// ---------------------------------------------------------------------------
extern "C" void kernel_launch(void* const* d_in, const int* in_sizes, int n_in,
                              void* d_out, int out_size, void* d_ws, size_t ws_size,
                              hipStream_t stream) {
    const float* z_seq     = (const float*)d_in[0];
    const float* act_seq   = (const float*)d_in[1];
    const float* goal_proj = (const float*)d_in[2];
    const float* h_init    = (const float*)d_in[3];
    const float* w_ih      = (const float*)d_in[4];
    const float* w_hh      = (const float*)d_in[5];
    const float* b_ih      = (const float*)d_in[6];
    const float* b_hh      = (const float*)d_in[7];
    float* out = (float*)d_out;

    unsigned short* wihb = (unsigned short*)d_ws;              // 768 * 576 bf16
    unsigned short* whhb = wihb + (size_t)G3 * WIH_STRIDE;     // 768 * 256 bf16
    // workspace use: 768*576*2 + 768*256*2 = 1,277,952 bytes

    pack_weights_kernel<<<G3, 128, 0, stream>>>(w_ih, w_hh, wihb, whhb);

    size_t lds_bytes =
        (size_t)(16 * XZ_STR + 16 * XA_STR + 16 * XZ_STR) * sizeof(unsigned short) +
        (size_t)(16 * F32_STR + 16 * PRE_STR + 16 * F32_STR + 16 * F32_STR) * sizeof(float);
    gru_scan_kernel<<<Bsz / 16, 512, lds_bytes, stream>>>(
        z_seq, act_seq, goal_proj, h_init, b_ih, b_hh, wihb, whhb, out);
}
